// LTAE_62027917689013
// MI455X (gfx1250) — matrix operand order repaired and checked by reading, not compiled, not run on hardware
//
#include <hip/hip_runtime.h>
#include <hip/hip_bf16.h>

// ---------------------------------------------------------------------------
// LTAE fused implementation for gfx1250 (MI455X), wave32 + WMMA f16.
//   K-proj GEMM, MLP GEMMs -> v_wmma_f32_16x16x32_f16
//   x (302MB fp32) read exactly once; e kept in LDS as f16.
//   MLP A-tiles staged via global_load_async_to_lds_b128 (ASYNCcnt).
// ---------------------------------------------------------------------------

#define DM     1536   // d_model
#define SEQL   24
#define NHEAD  16
#define EPITCH 1544   // f16 pitch of e_s rows (stride % 64 dwords == 4 -> no bank conflicts)
#define KSP    132    // f32 pitch of k_s rows
#define H1P    264    // f16 pitch of h1_s rows
#define H2P    132    // f32 pitch of h2_s rows
#define APITCH 40     // f16 pitch of async-staged A tile rows (20 dwords mod 64)

typedef __attribute__((ext_vector_type(16))) _Float16 v16h;
typedef __attribute__((ext_vector_type(8)))  float    v8f;
typedef __attribute__((ext_vector_type(4)))  _Float16 h4;

// Async copy of 16B global -> LDS, tracked with ASYNCcnt (CDNA5).
__device__ __forceinline__ void async_g2l_b128(const _Float16* gsrc,
                                               const _Float16* ldst) {
  asm volatile("global_load_async_to_lds_b128 %0, %1, off"
               :
               : "v"((unsigned)(size_t)ldst),          // LDS byte address (addr[31:0])
                 "v"((unsigned long long)(size_t)gsrc) // 64-bit global address
               : "memory");
}

// ---------------------------------------------------------------------------
// Pack fp32 weight matrix W[K][N] (row major) into f16 WMMA B-fragments:
// dst[((kb*NT + nt)*32 + lane)*16 + j]  with h = lane/16,
//   K = kb*32 + 8h + j (j<8), kb*32 + 16 + 8h + (j-8) (j>=8)
//   N = nt*16 + (lane&15)
// so each lane's 16 f16 are a single contiguous 32B load in the GEMM.
// ---------------------------------------------------------------------------
__global__ __launch_bounds__(256) void ltae_pack_b(const float* __restrict__ W,
                                                   _Float16* __restrict__ dst,
                                                   int N, int NT, int total) {
  const int idx = blockIdx.x * blockDim.x + threadIdx.x;
  if (idx >= total) return;
  const int lane = idx & 31;
  const int t    = idx >> 5;
  const int nti  = t % NT;
  const int kbi  = t / NT;
  const int nsrc = nti * 16 + (lane & 15);
  const int kbase = kbi * 32 + ((lane < 16) ? 0 : 8);
  _Float16* d = dst + (size_t)idx * 16;
#pragma unroll
  for (int j = 0; j < 16; ++j)
    d[j] = (_Float16)W[(size_t)(kbase + j + ((j < 8) ? 0 : 8)) * N + nsrc];
}

// Sinusoid positional table (rows 1..24 of the keras table, tiled over heads),
// stored f16 [24][1536].
__global__ __launch_bounds__(256) void ltae_pack_pos(_Float16* __restrict__ postab) {
  const int idx = blockIdx.x * blockDim.x + threadIdx.x;
  if (idx >= SEQL * DM) return;
  const int s = idx / DM;
  const int c = idx % DM;
  const int i = c % 96;                       // per-head width d_model/n_head
  const float ex = (float)(2 * (i / 2)) / 96.f;
  const float angle = (float)(s + 1) * powf(1000.f, -ex);
  const float v = (i & 1) ? cosf(angle) : sinf(angle);
  postab[idx] = (_Float16)v;
}

// ---------------------------------------------------------------------------
// Kernel A: one block per batch element.
//  LN(x)+pos -> e_s (f16, LDS)  ->  k = e@Wk (WMMA)  ->  softmax(Q.k)
//  -> o = attn . v  -> o_ws (f16)
// ---------------------------------------------------------------------------
__global__ __launch_bounds__(256) void ltae_attn_kernel(
    const float* __restrict__ x, const float* __restrict__ ln_g,
    const float* __restrict__ ln_b, const float* __restrict__ Q,
    const float* __restrict__ bk, const _Float16* __restrict__ Bpk,
    const _Float16* __restrict__ postab, _Float16* __restrict__ o_ws) {
  extern __shared__ __align__(16) char smem[];
  _Float16* e_s  = (_Float16*)smem;                           // 32 x EPITCH f16
  float*    k_s  = (float*)(smem + 32 * EPITCH * 2);          // 32 x KSP f32
  float*    attn = k_s + 32 * KSP;                            // 16 x 24 f32

  const int tid  = threadIdx.x;
  const int w    = tid >> 5;        // wave id (wave32)
  const int lane = tid & 31;
  const int b    = blockIdx.x;
  const float* xb = x + (size_t)b * SEQL * DM;

  // ---- Phase 1: LayerNorm + positional encoding, write e_s as f16 ----------
  for (int rr = 0; rr < 3; ++rr) {           // 8 waves x 3 rows = 24 rows
    const int s = w * 3 + rr;
    const float4* xrow = (const float4*)(xb + (size_t)s * DM);
    float4 xr[12];
    float sum = 0.f, ssq = 0.f;
#pragma unroll
    for (int t = 0; t < 12; ++t) {
      float4 v = xrow[lane + 32 * t];
      xr[t] = v;
      sum += v.x + v.y + v.z + v.w;
      ssq += v.x * v.x + v.y * v.y + v.z * v.z + v.w * v.w;
    }
#pragma unroll
    for (int off = 16; off > 0; off >>= 1) {
      sum += __shfl_xor(sum, off, 32);
      ssq += __shfl_xor(ssq, off, 32);
    }
    const float mean = sum * (1.f / (float)DM);
    const float rstd = rsqrtf(ssq * (1.f / (float)DM) - mean * mean + 1e-3f);
#pragma unroll
    for (int t = 0; t < 12; ++t) {
      const int c = 4 * (lane + 32 * t);
      const float4 g  = ((const float4*)ln_g)[lane + 32 * t];
      const float4 bb = ((const float4*)ln_b)[lane + 32 * t];
      const h4 p = *(const h4*)(postab + (size_t)s * DM + c);
      h4 ev;
      ev.x = (_Float16)((xr[t].x - mean) * rstd * g.x + bb.x + (float)p.x);
      ev.y = (_Float16)((xr[t].y - mean) * rstd * g.y + bb.y + (float)p.y);
      ev.z = (_Float16)((xr[t].z - mean) * rstd * g.z + bb.z + (float)p.z);
      ev.w = (_Float16)((xr[t].w - mean) * rstd * g.w + bb.w + (float)p.w);
      *(h4*)(e_s + (size_t)s * EPITCH + c) = ev;
    }
  }
  __syncthreads();

  // ---- Phase 2: k = e @ Wk + bk via WMMA (M=32 pad, N=128, K=1536) ---------
  {
    v8f acc0 = {}, acc1 = {};
    const int lrow = lane & 15;
    const int kA   = (lane < 16) ? 0 : 8;   // A-fragment K half select
    for (int kb = 0; kb < 48; ++kb) {
      const int k0 = kb * 32;
      v16h a0, a1, bf;
      ((uint4*)&a0)[0] = *(const uint4*)(e_s + (size_t)lrow * EPITCH + k0 + kA);
      ((uint4*)&a0)[1] = *(const uint4*)(e_s + (size_t)lrow * EPITCH + k0 + kA + 16);
      ((uint4*)&a1)[0] = *(const uint4*)(e_s + (size_t)(16 + lrow) * EPITCH + k0 + kA);
      ((uint4*)&a1)[1] = *(const uint4*)(e_s + (size_t)(16 + lrow) * EPITCH + k0 + kA + 16);
      const uint4* bp = (const uint4*)(Bpk + ((size_t)(kb * 8 + w) * 32 + lane) * 16);
      ((uint4*)&bf)[0] = bp[0];
      ((uint4*)&bf)[1] = bp[1];
      if (kb + 1 < 48)   // pull next B fragment into WGP-level caches
        __builtin_prefetch(Bpk + ((size_t)((kb + 1) * 8 + w) * 32 + lane) * 16, 0, 3);
      acc0 = __builtin_amdgcn_wmma_f32_16x16x32_f16(false, a0, false, bf,
                                                    (short)0, acc0, false, false);
      acc1 = __builtin_amdgcn_wmma_f32_16x16x32_f16(false, a1, false, bf,
                                                    (short)0, acc1, false, false);
    }
    const int nn   = w * 16 + lrow;         // global N column (wave w owns N-tile w)
    const float bv = bk[nn];
    const int mb   = (lane >> 4) * 8;
#pragma unroll
    for (int r = 0; r < 8; ++r) {
      k_s[(size_t)(mb + r) * KSP + nn]      = acc0[r] + bv;
      k_s[(size_t)(16 + mb + r) * KSP + nn] = acc1[r] + bv;
    }
  }
  __syncthreads();

  // ---- Phase 3: per-head scaled-dot softmax over S=24 ----------------------
  if (tid < NHEAD) {
    const int h = tid;
    float q[8];
#pragma unroll
    for (int j = 0; j < 8; ++j) q[j] = Q[h * 8 + j];
    float lg[SEQL];
    float mx = -1e30f;
    for (int s = 0; s < SEQL; ++s) {
      float a = 0.f;
#pragma unroll
      for (int j = 0; j < 8; ++j) a += q[j] * k_s[(size_t)s * KSP + h * 8 + j];
      a *= 0.35355339059f;  // 1/sqrt(8)
      lg[s] = a;
      mx = fmaxf(mx, a);
    }
    float sum = 0.f;
    for (int s = 0; s < SEQL; ++s) { float p = __expf(lg[s] - mx); lg[s] = p; sum += p; }
    const float inv = 1.f / sum;
    for (int s = 0; s < SEQL; ++s) attn[h * SEQL + s] = lg[s] * inv;
  }
  __syncthreads();

  // ---- Phase 4: o[c] = sum_s attn[h(c)][s] * e_s[s][c] ---------------------
#pragma unroll
  for (int i = 0; i < 6; ++i) {
    const int c = tid + 256 * i;
    const float* ah = attn + (c / 96) * SEQL;
    float acc = 0.f;
#pragma unroll
    for (int s = 0; s < SEQL; ++s)
      acc += ah[s] * (float)e_s[(size_t)s * EPITCH + c];
    o_ws[(size_t)b * DM + c] = (_Float16)acc;
  }
}

// ---------------------------------------------------------------------------
// Kernel B: batched MLP.  64 rows per block.
//  h1 = relu(bn1(o @ W1 + b1))  (WMMA, K=1536, N=256; A staged via async->LDS)
//  h2 = relu(bn2(h1 @ W2 + b2)) (WMMA, K=256,  N=128)
//  out = LN(h2)
// ---------------------------------------------------------------------------
__global__ __launch_bounds__(256) void ltae_mlp_kernel(
    const _Float16* __restrict__ o_ws, const _Float16* __restrict__ Bpk1,
    const _Float16* __restrict__ Bpk2, const float* __restrict__ b1,
    const float* __restrict__ bn1_g, const float* __restrict__ bn1_b,
    const float* __restrict__ bn1_m, const float* __restrict__ bn1_v,
    const float* __restrict__ b2, const float* __restrict__ bn2_g,
    const float* __restrict__ bn2_b, const float* __restrict__ bn2_m,
    const float* __restrict__ bn2_v, const float* __restrict__ out_g,
    const float* __restrict__ out_b, float* __restrict__ out) {
  extern __shared__ __align__(16) char smem[];
  _Float16* h1_s = (_Float16*)smem;                    // 64 x H1P f16
  float*    h2_s = (float*)(smem + 64 * H1P * 2);      // 64 x H2P f32 (GEMM2 epilogue)
  // Async A-tile staging buffers overlay the h2_s region (not live during GEMM1):
  _Float16* a_s  = (_Float16*)(smem + 64 * H1P * 2);   // 2 x 64 x APITCH f16 (10KB)

  const int tid  = threadIdx.x;
  const int w    = tid >> 5;
  const int lane = tid & 31;
  const int R    = blockIdx.x * 64;
  const int lrow = lane & 15;
  const int kA   = (lane < 16) ? 0 : 8;
  const int mb   = (lane >> 4) * 8;

  // Each thread owns one 16B chunk of the 64x32 f16 A tile per stage.
  const int srow   = tid >> 2;        // 0..63
  const int schunk = (tid & 3) * 8;   // f16 offset 0,8,16,24

  // ---- GEMM1: M=64, N=256, K=1536.  Wave w owns N-tiles {2w, 2w+1}. -------
  {
    v8f acc[4][2];
    const v8f z = {};
#pragma unroll
    for (int mt = 0; mt < 4; ++mt) { acc[mt][0] = z; acc[mt][1] = z; }

    // Prologue: stage K-block 0 into buffer 0 (ASYNCcnt-tracked).
    async_g2l_b128(o_ws + (size_t)(R + srow) * DM + schunk,
                   a_s + srow * APITCH + schunk);

    for (int kb = 0; kb < 48; ++kb) {
      const int cur = kb & 1;
      // All readers of the buffer we are about to overwrite have finished
      // (their LDS reads complete before their WMMAs, which precede this).
      __syncthreads();
      if (kb + 1 < 48) {
        async_g2l_b128(o_ws + (size_t)(R + srow) * DM + (kb + 1) * 32 + schunk,
                       a_s + (cur ^ 1) * (64 * APITCH) + srow * APITCH + schunk);
        asm volatile("s_wait_asynccnt 0x1" ::: "memory"); // current stage landed
      } else {
        asm volatile("s_wait_asynccnt 0x0" ::: "memory");
      }
      __syncthreads();  // every wave's staged data now LDS-visible

      const _Float16* abase = a_s + cur * (64 * APITCH);
      v16h af[4];
#pragma unroll
      for (int mt = 0; mt < 4; ++mt) {
        const _Float16* ap = abase + (size_t)(mt * 16 + lrow) * APITCH + kA;
        ((uint4*)&af[mt])[0] = *(const uint4*)ap;
        ((uint4*)&af[mt])[1] = *(const uint4*)(ap + 16);
      }
#pragma unroll
      for (int j = 0; j < 2; ++j) {
        v16h bf;
        const uint4* bp =
            (const uint4*)(Bpk1 + ((size_t)(kb * 16 + 2 * w + j) * 32 + lane) * 16);
        ((uint4*)&bf)[0] = bp[0];
        ((uint4*)&bf)[1] = bp[1];
        if (kb + 1 < 48)
          __builtin_prefetch(
              Bpk1 + ((size_t)((kb + 1) * 16 + 2 * w + j) * 32 + lane) * 16, 0, 3);
#pragma unroll
        for (int mt = 0; mt < 4; ++mt)
          acc[mt][j] = __builtin_amdgcn_wmma_f32_16x16x32_f16(
              false, af[mt], false, bf, (short)0, acc[mt][j], false, false);
      }
    }
    // Fused bias + BN + ReLU epilogue -> h1_s (f16)
#pragma unroll
    for (int j = 0; j < 2; ++j) {
      const int n = (2 * w + j) * 16 + lrow;
      const float s1  = bn1_g[n] * rsqrtf(bn1_v[n] + 1e-3f);
      const float sh1 = bn1_b[n] - bn1_m[n] * s1;
      const float bb  = b1[n];
#pragma unroll
      for (int mt = 0; mt < 4; ++mt) {
        const int m0 = mt * 16 + mb;
#pragma unroll
        for (int r = 0; r < 8; ++r) {
          const float v = (acc[mt][j][r] + bb) * s1 + sh1;
          h1_s[(size_t)(m0 + r) * H1P + n] = (_Float16)fmaxf(v, 0.f);
        }
      }
    }
  }
  __syncthreads();

  // ---- GEMM2: M=64, N=128, K=256.  Wave w owns N-tile w. -------------------
  {
    v8f acc[4];
    const v8f z = {};
#pragma unroll
    for (int mt = 0; mt < 4; ++mt) acc[mt] = z;
    for (int kb = 0; kb < 8; ++kb) {
      v16h bf;
      const uint4* bp = (const uint4*)(Bpk2 + ((size_t)(kb * 8 + w) * 32 + lane) * 16);
      ((uint4*)&bf)[0] = bp[0];
      ((uint4*)&bf)[1] = bp[1];
#pragma unroll
      for (int mt = 0; mt < 4; ++mt) {
        v16h af;
        const _Float16* ap = h1_s + (size_t)(mt * 16 + lrow) * H1P + kb * 32 + kA;
        ((uint4*)&af)[0] = *(const uint4*)ap;
        ((uint4*)&af)[1] = *(const uint4*)(ap + 16);
        acc[mt] = __builtin_amdgcn_wmma_f32_16x16x32_f16(
            false, af, false, bf, (short)0, acc[mt], false, false);
      }
    }
    const int n = w * 16 + lrow;
    const float s2  = bn2_g[n] * rsqrtf(bn2_v[n] + 1e-3f);
    const float sh2 = bn2_b[n] - bn2_m[n] * s2;
    const float bb  = b2[n];
#pragma unroll
    for (int mt = 0; mt < 4; ++mt)
#pragma unroll
      for (int r = 0; r < 8; ++r)
        h2_s[(size_t)(mt * 16 + mb + r) * H2P + n] =
            fmaxf((acc[mt][r] + bb) * s2 + sh2, 0.f);
  }
  __syncthreads();

  // ---- Final LayerNorm over 128 channels; 8 rows per wave ------------------
#pragma unroll
  for (int i = 0; i < 8; ++i) {
    const int m = w * 8 + i;
    const float4 v = ((const float4*)(h2_s + (size_t)m * H2P))[lane];
    float sum = v.x + v.y + v.z + v.w;
    float ssq = v.x * v.x + v.y * v.y + v.z * v.z + v.w * v.w;
#pragma unroll
    for (int off = 16; off > 0; off >>= 1) {
      sum += __shfl_xor(sum, off, 32);
      ssq += __shfl_xor(ssq, off, 32);
    }
    const float mean = sum * (1.f / 128.f);
    const float rstd = rsqrtf(ssq * (1.f / 128.f) - mean * mean + 1e-3f);
    const float4 g  = ((const float4*)out_g)[lane];
    const float4 bo = ((const float4*)out_b)[lane];
    float4 o;
    o.x = (v.x - mean) * rstd * g.x + bo.x;
    o.y = (v.y - mean) * rstd * g.y + bo.y;
    o.z = (v.z - mean) * rstd * g.z + bo.z;
    o.w = (v.w - mean) * rstd * g.w + bo.w;
    ((float4*)(out + (size_t)(R + m) * 128))[lane] = o;
  }
}

// ---------------------------------------------------------------------------
extern "C" void kernel_launch(void* const* d_in, const int* in_sizes, int n_in,
                              void* d_out, int out_size, void* d_ws, size_t ws_size,
                              hipStream_t stream) {
  (void)in_sizes; (void)n_in; (void)out_size; (void)ws_size;
  const float* x     = (const float*)d_in[0];
  const float* ln_g  = (const float*)d_in[1];
  const float* ln_b  = (const float*)d_in[2];
  const float* Q     = (const float*)d_in[3];
  const float* Wk    = (const float*)d_in[4];
  const float* bk    = (const float*)d_in[5];
  const float* W1    = (const float*)d_in[6];
  const float* b1    = (const float*)d_in[7];
  const float* bn1_g = (const float*)d_in[8];
  const float* bn1_b = (const float*)d_in[9];
  const float* bn1_m = (const float*)d_in[10];
  const float* bn1_v = (const float*)d_in[11];
  const float* W2    = (const float*)d_in[12];
  const float* b2    = (const float*)d_in[13];
  const float* bn2_g = (const float*)d_in[14];
  const float* bn2_b = (const float*)d_in[15];
  const float* bn2_m = (const float*)d_in[16];
  const float* bn2_v = (const float*)d_in[17];
  const float* out_g = (const float*)d_in[18];
  const float* out_b = (const float*)d_in[19];
  float* out = (float*)d_out;

  char* ws = (char*)d_ws;
  _Float16* Bpk_k  = (_Float16*)(ws + 0);        //  48*8*32*16  f16 = 393216 B
  _Float16* Bpk_1  = (_Float16*)(ws + 393216);   //  48*16*32*16 f16 = 786432 B
  _Float16* Bpk_2  = (_Float16*)(ws + 1179648);  //  8*8*32*16   f16 =  65536 B
  _Float16* postab = (_Float16*)(ws + 1245184);  //  24*1536     f16 =  73728 B
  _Float16* o_ws   = (_Float16*)(ws + 1318912);  //  2048*1536   f16 = 6291456 B

  // Weight packing + positional table (small; L2 resident afterwards)
  ltae_pack_b<<<48, 256, 0, stream>>>(Wk, Bpk_k, 128, 8, 48 * 8 * 32);
  ltae_pack_b<<<96, 256, 0, stream>>>(W1, Bpk_1, 256, 16, 48 * 16 * 32);
  ltae_pack_b<<<8, 256, 0, stream>>>(W2, Bpk_2, 128, 8, 8 * 8 * 32);
  ltae_pack_pos<<<144, 256, 0, stream>>>(postab);

  // Fused LN + pos + K-proj (WMMA) + attention -> o (f16)
  const int smemA = 32 * EPITCH * 2 + 32 * KSP * 4 + NHEAD * SEQL * 4; // 117248 B
  ltae_attn_kernel<<<2048, 256, smemA, stream>>>(x, ln_g, ln_b, Q, bk, Bpk_k,
                                                 postab, o_ws);

  // Batched MLP (WMMA, async A-tile staging) + BN/ReLU + final LN
  const int smemB = 64 * H1P * 2 + 64 * H2P * 4; // 67584 B
  ltae_mlp_kernel<<<32, 256, smemB, stream>>>(o_ws, Bpk_1, Bpk_2, b1, bn1_g,
                                              bn1_b, bn1_m, bn1_v, b2, bn2_g,
                                              bn2_b, bn2_m, bn2_v, out_g, out_b,
                                              out);
}
